// GRU_35510789604045
// MI455X (gfx1250) — matrix operand order repaired and checked by reading, not compiled, not run on hardware
//
#include <hip/hip_runtime.h>

// ---------------- problem sizes ----------------
#define S_T     1000
#define B_B     64
#define D_IN    440
#define D_P     448      // D padded to multiple of 32 (WMMA K-step)
#define H_H     512
#define N_G     1536     // 3*H (gate order: h, z, r)
#define O_O     1000
#define O_P     1024     // O padded to multiple of 16
#define M_ROWS  64000    // S*B
#define REC_BLOCKS 16
#define BN_EPS  1e-5f

// ---------------- WMMA types (wave32, gfx1250) ----------------
typedef __attribute__((ext_vector_type(16))) __bf16        v16bf;
typedef __attribute__((ext_vector_type(8)))  float         v8f;
typedef __attribute__((ext_vector_type(4)))  unsigned int  v4u;

union Frag16 { v16bf v; unsigned int w[8]; };

__device__ __forceinline__ unsigned short f2bf(float f) {
  unsigned int u = __float_as_uint(f);
  return (unsigned short)((u + 0x7FFFu + ((u >> 16) & 1u)) >> 16);  // RNE
}
__device__ __forceinline__ float bf2f(unsigned short h) {
  return __uint_as_float(((unsigned int)h) << 16);
}

// A fragment 16x32 bf16 from row-major X[m][k] (ld in elements).
// lane L (L%16 = m-offset): K = 16*hi + 0..15  (hi = lane>=16)
__device__ __forceinline__ v16bf load_a_frag(const unsigned short* base, int ld,
                                             int r0, int k0, int lane) {
  const unsigned short* p = base + (size_t)(r0 + (lane & 15)) * ld
                                 + (k0 + ((lane >> 4) & 1) * 16);
  Frag16 f;
  *(v4u*)(&f.w[0]) = *(const v4u*)(p);
  *(v4u*)(&f.w[4]) = *(const v4u*)(p + 8);
  return f.v;
}
// B fragment 32x16 bf16 taken from row-major W[n][k]: lane holds col n = n0+lane%16,
// K = 16*hi + 0..15 -> one contiguous 32B run of W's row n.
__device__ __forceinline__ v16bf load_b_frag(const unsigned short* base, int ld,
                                             int n0, int k0, int lane) {
  const unsigned short* p = base + (size_t)(n0 + (lane & 15)) * ld
                                 + (k0 + ((lane >> 4) & 1) * 16);
  Frag16 f;
  *(v4u*)(&f.w[0]) = *(const v4u*)(p);
  *(v4u*)(&f.w[4]) = *(const v4u*)(p + 8);
  return f.v;
}
// Same as load_b_frag but from an LDS slice with row stride 512 (ds_load_b128 path).
__device__ __forceinline__ v16bf load_b_frag_lds(const unsigned short* base,
                                                 int n0, int k0, int lane) {
  const unsigned short* p = base + (n0 + (lane & 15)) * 512
                                 + (k0 + ((lane >> 4) & 1) * 16);
  Frag16 f;
  *(v4u*)(&f.w[0]) = *(const v4u*)(p);
  *(v4u*)(&f.w[4]) = *(const v4u*)(p + 8);
  return f.v;
}

__device__ __forceinline__ v8f wmma_bf16(v16bf a, v16bf b, v8f c) {
  // (neg_a, A, neg_b, B, c_mod, C, reuse_a, reuse_b)
  return __builtin_amdgcn_wmma_f32_16x16x32_bf16(false, a, false, b, (short)0, c,
                                                 false, false);
}

// ---------------- device-scope barrier (persistent recurrence kernel) --------
__device__ __forceinline__ void grid_barrier(int* cnt, int* rel, int gen, int nblk) {
  __syncthreads();
  if (threadIdx.x == 0) {
    __threadfence();
    int prev = __hip_atomic_fetch_add(cnt, 1, __ATOMIC_ACQ_REL, __HIP_MEMORY_SCOPE_AGENT);
    if (prev == gen * nblk + (nblk - 1)) {
      __hip_atomic_store(rel, gen + 1, __ATOMIC_RELEASE, __HIP_MEMORY_SCOPE_AGENT);
    } else {
      while (__hip_atomic_load(rel, __ATOMIC_ACQUIRE, __HIP_MEMORY_SCOPE_AGENT) < gen + 1)
        __builtin_amdgcn_s_sleep(2);
    }
  }
  __syncthreads();
  __threadfence();
}

// ---------------- packing / init kernels ----------------
__global__ void k_pack_x(const float* __restrict__ x, unsigned short* __restrict__ xb,
                         int total) {
  int i = blockIdx.x * blockDim.x + threadIdx.x;
  if (i >= total) return;
  int m = i / D_P, k = i - m * D_P;
  xb[i] = (k < D_IN) ? f2bf(x[(size_t)m * D_IN + k]) : (unsigned short)0;
}

__global__ void k_pack_w(const float* __restrict__ Wh, const float* __restrict__ Wz,
                         const float* __restrict__ Wr, unsigned short* __restrict__ wb,
                         int total) {
  int i = blockIdx.x * blockDim.x + threadIdx.x;
  if (i >= total) return;
  int n = i / D_P, k = i - n * D_P;
  int g = n >> 9, h = n & 511;
  const float* src = (g == 0) ? Wh : (g == 1) ? Wz : Wr;
  wb[i] = (k < D_IN) ? f2bf(src[(size_t)h * D_IN + k]) : (unsigned short)0;
}

__global__ void k_pack_u(const float* __restrict__ Uh, const float* __restrict__ Uz,
                         const float* __restrict__ Ur, unsigned short* __restrict__ ub,
                         int total) {
  int i = blockIdx.x * blockDim.x + threadIdx.x;
  if (i >= total) return;
  int g = i >> 18, r = i & 262143;   // 512*512 per gate, order [Uh,Uz,Ur]
  const float* src = (g == 0) ? Uh : (g == 1) ? Uz : Ur;
  ub[i] = f2bf(src[r]);
}

__global__ void k_pack_wf(const float* __restrict__ Wf, unsigned short* __restrict__ wfb,
                          int total) {
  int i = blockIdx.x * blockDim.x + threadIdx.x;
  if (i >= total) return;
  int n = i >> 9, k = i & 511;
  wfb[i] = (n < O_O) ? f2bf(Wf[(size_t)n * H_H + k]) : (unsigned short)0;
}

__global__ void k_init(float* __restrict__ hcur, unsigned short* __restrict__ hbf,
                       int* __restrict__ bar) {
  int i = blockIdx.x * blockDim.x + threadIdx.x;
  if (i < B_B * H_H) { hcur[i] = 0.f; hbf[i] = 0; }
  if (i < 2) bar[i] = 0;
}

// ---------------- GEMM1: gates_raw = x_bf16 @ Wall^T, + column moments --------
// grid: (N_G/128, M_ROWS/64), block 256 (8 waves).
// Register blocking: each wave owns a 32x32 tile (2 A-frags x 2 B-frags ->
// 4 WMMAs per k-step); consecutive WMMAs reuse loaded fragments so the matrix
// pipe overlaps the next k-step's global_load_b128 clauses.
__global__ __launch_bounds__(256)
void k_gemm_gates(const unsigned short* __restrict__ xb, const unsigned short* __restrict__ wb,
                  unsigned short* __restrict__ gates,
                  float* __restrict__ pS, float* __restrict__ pQ) {
  __shared__ float psum[8][32];
  __shared__ float psq[8][32];
  const int tid = threadIdx.x, lane = tid & 31, w = tid >> 5;
  const int wm = w & 1, wn = w >> 1;                 // 2x4 wave grid
  const int m0 = blockIdx.y * 64 + wm * 32;          // m-tiles at +0, +16
  const int n0 = blockIdx.x * 128 + wn * 32;         // n-tiles at +0, +16

  v8f acc[2][2];
  #pragma unroll
  for (int mi = 0; mi < 2; ++mi)
    #pragma unroll
    for (int ni = 0; ni < 2; ++ni)
      #pragma unroll
      for (int j = 0; j < 8; ++j) acc[mi][ni][j] = 0.f;

  constexpr int KS = D_P / 32;
  v16bf a[2][2], b[2][2];                            // [buf][tile]
  a[0][0] = load_a_frag(xb, D_P, m0, 0, lane);
  a[0][1] = load_a_frag(xb, D_P, m0 + 16, 0, lane);
  b[0][0] = load_b_frag(wb, D_P, n0, 0, lane);
  b[0][1] = load_b_frag(wb, D_P, n0 + 16, 0, lane);
  #pragma unroll
  for (int kk = 0; kk < KS; ++kk) {
    const int cur = kk & 1, nxt = cur ^ 1;
    if (kk + 1 < KS) {
      const int k0 = (kk + 1) * 32;
      a[nxt][0] = load_a_frag(xb, D_P, m0, k0, lane);
      a[nxt][1] = load_a_frag(xb, D_P, m0 + 16, k0, lane);
      b[nxt][0] = load_b_frag(wb, D_P, n0, k0, lane);
      b[nxt][1] = load_b_frag(wb, D_P, n0 + 16, k0, lane);
    }
    acc[0][0] = wmma_bf16(a[cur][0], b[cur][0], acc[0][0]);
    acc[1][0] = wmma_bf16(a[cur][1], b[cur][0], acc[1][0]);
    acc[0][1] = wmma_bf16(a[cur][0], b[cur][1], acc[0][1]);
    acc[1][1] = wmma_bf16(a[cur][1], b[cur][1], acc[1][1]);
  }

  const int mhalf = ((lane >> 4) & 1) * 8;
  const int t = blockIdx.y;          // block covers rows t*64 .. t*64+63
  #pragma unroll
  for (int ni = 0; ni < 2; ++ni) {
    const int n = n0 + ni * 16 + (lane & 15);
    const int g = n >> 9, h = n & 511;
    float s = 0.f, q = 0.f;
    #pragma unroll
    for (int mi = 0; mi < 2; ++mi) {
      #pragma unroll
      for (int j = 0; j < 8; ++j) {
        const int bb = wm * 32 + mi * 16 + j + mhalf;   // row within the 64-block
        float v = acc[mi][ni][j];
        gates[(((size_t)t * 3 + g) * 64 + bb) * 512 + h] = f2bf(v);
        s += v; q += v * v;
      }
    }
    s += __shfl_xor(s, 16, 32);      // combine the two m-halves (same n)
    q += __shfl_xor(q, 16, 32);
    if (lane < 16) { psum[w][ni * 16 + lane] = s; psq[w][ni * 16 + lane] = q; }
  }
  __syncthreads();
  if (tid < 128) {                   // fixed-order combine: waves 2*wn, 2*wn+1
    const int nl = tid, wbase = (nl >> 5) * 2, idx = nl & 31;
    float s = psum[wbase][idx] + psum[wbase + 1][idx];
    float q = psq[wbase][idx] + psq[wbase + 1][idx];
    const int n = blockIdx.x * 128 + nl;
    pS[(size_t)blockIdx.y * N_G + n] = s;
    pQ[(size_t)blockIdx.y * N_G + n] = q;
  }
}

// ---------------- BN stats -> affine fold (a[n], c[n]) ----------------
__global__ void k_stats_gates(const float* __restrict__ pS, const float* __restrict__ pQ,
                              const float* gh, const float* bh, const float* gz, const float* bz,
                              const float* gr, const float* br,
                              float* __restrict__ a1, float* __restrict__ c1) {
  int n = blockIdx.x * blockDim.x + threadIdx.x;
  if (n >= N_G) return;
  float s = 0.f, q = 0.f;
  for (int b = 0; b < M_ROWS / 64; ++b) { s += pS[(size_t)b * N_G + n]; q += pQ[(size_t)b * N_G + n]; }
  float mean = s * (1.f / (float)M_ROWS);
  float var  = q * (1.f / (float)M_ROWS) - mean * mean;
  int g = n >> 9, h = n & 511;
  float gam = (g == 0) ? gh[h] : (g == 1) ? gz[h] : gr[h];
  float bet = (g == 0) ? bh[h] : (g == 1) ? bz[h] : br[h];
  float sc = gam * rsqrtf(var + BN_EPS);
  a1[n] = sc;
  c1[n] = bet - mean * sc;
}

__global__ void k_stats_final(const float* __restrict__ pS, const float* __restrict__ pQ,
                              const float* gf, const float* bf_,
                              float* __restrict__ a2, float* __restrict__ c2) {
  int n = blockIdx.x * blockDim.x + threadIdx.x;
  if (n >= O_O) return;
  float s = 0.f, q = 0.f;
  for (int b = 0; b < M_ROWS / 64; ++b) { s += pS[(size_t)b * O_P + n]; q += pQ[(size_t)b * O_P + n]; }
  float mean = s * (1.f / (float)M_ROWS);
  float var  = q * (1.f / (float)M_ROWS) - mean * mean;
  float sc = gf[n] * rsqrtf(var + BN_EPS);
  a2[n] = sc;
  c2[n] = bf_[n] - mean * sc;
}

// ---------------- persistent GRU recurrence ----------------
// 16 blocks x 256 threads. Block nb owns H-columns [nb*32, nb*32+32).
// U slices for this block (3 gates x 32 rows x 512 bf16 = 96KB) are staged into
// LDS once and re-used for all 1000 steps (B fragments come from ds_load_b128).
// Per step: phase1 z/r (WMMA over h_bf16), write r*h; grid barrier;
//           phase2 hcand (WMMA over r*h), update h; grid barrier.
__global__ __launch_bounds__(256)
void k_recurrence(const unsigned short* __restrict__ gates,
                  const unsigned short* __restrict__ ub,   // [Uh|Uz|Ur] bf16, 512x512 each
                  const float* __restrict__ a1, const float* __restrict__ c1,
                  float* __restrict__ hcur, unsigned short* __restrict__ hbf,
                  unsigned short* __restrict__ rh, unsigned short* __restrict__ hall,
                  int* __restrict__ bar) {
  extern __shared__ unsigned short smem[];   // 3*32*512 bf16 = 98304 B
  const int tid = threadIdx.x, lane = tid & 31, w = tid >> 5;
  const int m0  = (w & 3) * 16;                      // batch rows (B=64 -> 4 tiles)
  const int n0l = (w >> 2) * 16;                     // local H columns within slice
  const int mhalf = ((lane >> 4) & 1) * 8;
  const int nn = blockIdx.x * 32 + n0l + (lane & 15);

  // stage this block's U rows into LDS (once)
  for (int i = tid; i < 3 * 32 * 64; i += 256) {     // v4u granules (16B)
    const int g = i >> 11;                           // 32*64 = 2048 per gate
    const int rem = i & 2047;
    const int row = rem >> 6, c4 = rem & 63;
    const v4u* src = (const v4u*)(ub + ((size_t)(g * 512 + blockIdx.x * 32 + row)) * 512) + c4;
    ((v4u*)smem)[i] = *src;
  }
  __syncthreads();
  const unsigned short* sUh = smem;
  const unsigned short* sUz = smem + 32 * 512;
  const unsigned short* sUr = smem + 64 * 512;

  const float a1h = a1[nn],        c1h = c1[nn];
  const float a1z = a1[512 + nn],  c1z = c1[512 + nn];
  const float a1r = a1[1024 + nn], c1r = c1[1024 + nn];

  for (int t = 0; t < S_T; ++t) {
    const size_t base_t = (size_t)t * 3 * 32768;     // 32768 = 64*512
    // prefetch this step's z/r gate rows while the WMMA chain runs
    #pragma unroll
    for (int j = 0; j < 8; ++j) {
      const size_t o = (size_t)(m0 + j + mhalf) * 512 + nn;
      __builtin_prefetch(&gates[base_t + 32768 + o], 0, 3);
      __builtin_prefetch(&gates[base_t + 65536 + o], 0, 3);
    }
    // ---- phase 1: z, r ----
    v8f az, ar;
    #pragma unroll
    for (int j = 0; j < 8; ++j) { az[j] = 0.f; ar[j] = 0.f; }
    {
      v16bf acur = load_a_frag(hbf, 512, m0, 0, lane);
      #pragma unroll
      for (int kk = 0; kk < 16; ++kk) {
        v16bf anxt = acur;
        if (kk < 15) anxt = load_a_frag(hbf, 512, m0, (kk + 1) * 32, lane);
        const int k0 = kk * 32;
        az = wmma_bf16(acur, load_b_frag_lds(sUz, n0l, k0, lane), az);
        ar = wmma_bf16(acur, load_b_frag_lds(sUr, n0l, k0, lane), ar);
        acur = anxt;
      }
    }
    v8f zt, ho;
    #pragma unroll
    for (int j = 0; j < 8; ++j) {
      const int b = m0 + j + mhalf;
      const size_t o = (size_t)b * 512 + nn;
      float pz = bf2f(gates[base_t + 32768 + o]) * a1z + c1z + az[j];
      float pr = bf2f(gates[base_t + 65536 + o]) * a1r + c1r + ar[j];
      float z = 1.f / (1.f + __expf(-pz));
      float r = 1.f / (1.f + __expf(-pr));
      float h0v = hcur[o];
      zt[j] = z; ho[j] = h0v;
      rh[o] = f2bf(r * h0v);
    }
    grid_barrier(bar, bar + 1, 2 * t, REC_BLOCKS);
    // prefetch h-gate rows for the epilogue
    #pragma unroll
    for (int j = 0; j < 8; ++j)
      __builtin_prefetch(&gates[base_t + (size_t)(m0 + j + mhalf) * 512 + nn], 0, 3);
    // ---- phase 2: hcand and state update ----
    v8f ah;
    #pragma unroll
    for (int j = 0; j < 8; ++j) ah[j] = 0.f;
    {
      v16bf acur = load_a_frag(rh, 512, m0, 0, lane);
      #pragma unroll
      for (int kk = 0; kk < 16; ++kk) {
        v16bf anxt = acur;
        if (kk < 15) anxt = load_a_frag(rh, 512, m0, (kk + 1) * 32, lane);
        ah = wmma_bf16(acur, load_b_frag_lds(sUh, n0l, kk * 32, lane), ah);
        acur = anxt;
      }
    }
    #pragma unroll
    for (int j = 0; j < 8; ++j) {
      const int b = m0 + j + mhalf;
      const size_t o = (size_t)b * 512 + nn;
      float ph = bf2f(gates[base_t + o]) * a1h + c1h + ah[j];
      float hc = tanhf(ph);
      float hn = zt[j] * ho[j] + (1.f - zt[j]) * hc;
      hcur[o] = hn;
      unsigned short hb = f2bf(hn);
      hbf[o] = hb;
      hall[(size_t)t * 32768 + o] = hb;
    }
    grid_barrier(bar, bar + 1, 2 * t + 1, REC_BLOCKS);
  }
}

// ---------------- GEMM3: ws_raw = h_all @ Wf^T (into d_out) + column moments --
__global__ __launch_bounds__(256)
void k_gemm_final(const unsigned short* __restrict__ hall, const unsigned short* __restrict__ wfb,
                  float* __restrict__ out, float* __restrict__ pS, float* __restrict__ pQ) {
  __shared__ float psum[8][32];
  __shared__ float psq[8][32];
  const int tid = threadIdx.x, lane = tid & 31, w = tid >> 5;
  const int wm = w & 1, wn = w >> 1;
  const int m0 = blockIdx.y * 64 + wm * 32;
  const int n0 = blockIdx.x * 128 + wn * 32;

  v8f acc[2][2];
  #pragma unroll
  for (int mi = 0; mi < 2; ++mi)
    #pragma unroll
    for (int ni = 0; ni < 2; ++ni)
      #pragma unroll
      for (int j = 0; j < 8; ++j) acc[mi][ni][j] = 0.f;

  constexpr int KS = H_H / 32;
  v16bf a[2][2], b[2][2];
  a[0][0] = load_a_frag(hall, H_H, m0, 0, lane);
  a[0][1] = load_a_frag(hall, H_H, m0 + 16, 0, lane);
  b[0][0] = load_b_frag(wfb, H_H, n0, 0, lane);
  b[0][1] = load_b_frag(wfb, H_H, n0 + 16, 0, lane);
  #pragma unroll
  for (int kk = 0; kk < KS; ++kk) {
    const int cur = kk & 1, nxt = cur ^ 1;
    if (kk + 1 < KS) {
      const int k0 = (kk + 1) * 32;
      a[nxt][0] = load_a_frag(hall, H_H, m0, k0, lane);
      a[nxt][1] = load_a_frag(hall, H_H, m0 + 16, k0, lane);
      b[nxt][0] = load_b_frag(wfb, H_H, n0, k0, lane);
      b[nxt][1] = load_b_frag(wfb, H_H, n0 + 16, k0, lane);
    }
    acc[0][0] = wmma_bf16(a[cur][0], b[cur][0], acc[0][0]);
    acc[1][0] = wmma_bf16(a[cur][1], b[cur][0], acc[1][0]);
    acc[0][1] = wmma_bf16(a[cur][0], b[cur][1], acc[0][1]);
    acc[1][1] = wmma_bf16(a[cur][1], b[cur][1], acc[1][1]);
  }

  const int mhalf = ((lane >> 4) & 1) * 8;
  #pragma unroll
  for (int ni = 0; ni < 2; ++ni) {
    const int n = n0 + ni * 16 + (lane & 15);
    float s = 0.f, q = 0.f;
    #pragma unroll
    for (int mi = 0; mi < 2; ++mi) {
      #pragma unroll
      for (int j = 0; j < 8; ++j) {
        const int m = m0 + mi * 16 + j + mhalf;
        float v = acc[mi][ni][j];
        if (n < O_O) out[(size_t)m * O_O + n] = v;
        s += v; q += v * v;
      }
    }
    s += __shfl_xor(s, 16, 32);
    q += __shfl_xor(q, 16, 32);
    if (lane < 16) { psum[w][ni * 16 + lane] = s; psq[w][ni * 16 + lane] = q; }
  }
  __syncthreads();
  if (tid < 128) {
    const int nl = tid, wbase = (nl >> 5) * 2, idx = nl & 31;
    float s = psum[wbase][idx] + psum[wbase + 1][idx];
    float q = psq[wbase][idx] + psq[wbase + 1][idx];
    const int n = blockIdx.x * 128 + nl;
    pS[(size_t)blockIdx.y * O_P + n] = s;
    pQ[(size_t)blockIdx.y * O_P + n] = q;
  }
}

// ---------------- fused BN + log_softmax, one block per row ----------------
__global__ __launch_bounds__(256)
void k_bn_logsoftmax(float* __restrict__ out, const float* __restrict__ a2,
                     const float* __restrict__ c2) {
  __shared__ float vals[O_O];
  __shared__ float red[256];
  const int tid = threadIdx.x;
  float* row = out + (size_t)blockIdx.x * O_O;
  float mx = -3.402823466e38f;
  for (int n = tid; n < O_O; n += 256) {
    float v = row[n] * a2[n] + c2[n];
    vals[n] = v;
    mx = fmaxf(mx, v);
  }
  red[tid] = mx; __syncthreads();
  for (int s = 128; s > 0; s >>= 1) {
    if (tid < s) red[tid] = fmaxf(red[tid], red[tid + s]);
    __syncthreads();
  }
  mx = red[0]; __syncthreads();
  float sm = 0.f;
  for (int n = tid; n < O_O; n += 256) sm += __expf(vals[n] - mx);
  red[tid] = sm; __syncthreads();
  for (int s = 128; s > 0; s >>= 1) {
    if (tid < s) red[tid] += red[tid + s];
    __syncthreads();
  }
  float lse = __logf(red[0]);
  for (int n = tid; n < O_O; n += 256) row[n] = vals[n] - mx - lse;
}

// ---------------- launch ----------------
extern "C" void kernel_launch(void* const* d_in, const int* in_sizes, int n_in,
                              void* d_out, int out_size, void* d_ws, size_t ws_size,
                              hipStream_t stream) {
  (void)in_sizes; (void)n_in; (void)out_size; (void)ws_size;
  const float* x  = (const float*)d_in[0];
  const float* Wh = (const float*)d_in[1];
  const float* Wz = (const float*)d_in[2];
  const float* Wr = (const float*)d_in[3];
  const float* Uh = (const float*)d_in[4];
  const float* Uz = (const float*)d_in[5];
  const float* Ur = (const float*)d_in[6];
  const float* gh = (const float*)d_in[7];
  const float* bh = (const float*)d_in[8];
  const float* gz = (const float*)d_in[9];
  const float* bz = (const float*)d_in[10];
  const float* gr = (const float*)d_in[11];
  const float* br = (const float*)d_in[12];
  const float* Wf = (const float*)d_in[13];
  const float* gf = (const float*)d_in[14];
  const float* bff = (const float*)d_in[15];
  float* out = (float*)d_out;

  // workspace carve (256B aligned)
  char* p = (char*)d_ws;
  auto take = [&](size_t bytes) -> char* {
    char* r = p; p += (bytes + 255) & ~(size_t)255; return r;
  };
  unsigned short* xb    = (unsigned short*)take((size_t)M_ROWS * D_P * 2);   // 57.3 MB
  unsigned short* wb    = (unsigned short*)take((size_t)N_G * D_P * 2);
  unsigned short* ub    = (unsigned short*)take((size_t)3 * 512 * 512 * 2);
  unsigned short* wfb   = (unsigned short*)take((size_t)O_P * H_H * 2);
  unsigned short* gates = (unsigned short*)take((size_t)S_T * 3 * B_B * H_H * 2); // 197 MB
  unsigned short* hall  = (unsigned short*)take((size_t)S_T * B_B * H_H * 2);     // 65.5 MB
  float*          hcur  = (float*)take((size_t)B_B * H_H * 4);
  unsigned short* hbf   = (unsigned short*)take((size_t)B_B * H_H * 2);
  unsigned short* rh    = (unsigned short*)take((size_t)B_B * H_H * 2);
  float* pS1 = (float*)take((size_t)(M_ROWS / 64) * N_G * 4);
  float* pQ1 = (float*)take((size_t)(M_ROWS / 64) * N_G * 4);
  float* pS2 = (float*)take((size_t)(M_ROWS / 64) * O_P * 4);
  float* pQ2 = (float*)take((size_t)(M_ROWS / 64) * O_P * 4);
  float* a1  = (float*)take((size_t)N_G * 4);
  float* c1  = (float*)take((size_t)N_G * 4);
  float* a2  = (float*)take((size_t)O_P * 4);
  float* c2  = (float*)take((size_t)O_P * 4);
  int*   bar = (int*)take(256);

  // 0) state init (h0 = 0, barrier counters)
  k_init<<<dim3(128), dim3(256), 0, stream>>>(hcur, hbf, bar);
  // 1) pack bf16 operands
  {
    int t1 = M_ROWS * D_P;
    k_pack_x<<<dim3((t1 + 255) / 256), dim3(256), 0, stream>>>(x, xb, t1);
    int t2 = N_G * D_P;
    k_pack_w<<<dim3((t2 + 255) / 256), dim3(256), 0, stream>>>(Wh, Wz, Wr, wb, t2);
    int t3 = 3 * 512 * 512;
    k_pack_u<<<dim3((t3 + 255) / 256), dim3(256), 0, stream>>>(Uh, Uz, Ur, ub, t3);
    int t4 = O_P * H_H;
    k_pack_wf<<<dim3((t4 + 255) / 256), dim3(256), 0, stream>>>(Wf, wfb, t4);
  }
  // 2) gate projections (raw) + column moments
  k_gemm_gates<<<dim3(N_G / 128, M_ROWS / 64), dim3(256), 0, stream>>>(xb, wb, gates, pS1, pQ1);
  // 3) fold BN into affine a1/c1
  k_stats_gates<<<dim3((N_G + 255) / 256), dim3(256), 0, stream>>>(pS1, pQ1, gh, bh, gz, bz, gr, br, a1, c1);
  // 4) sequential GRU recurrence (persistent, grid-synchronized, U in LDS)
  k_recurrence<<<dim3(REC_BLOCKS), dim3(256), 98304, stream>>>(gates, ub, a1, c1, hcur, hbf, rh, hall, bar);
  // 5) output projection (raw -> d_out) + column moments
  k_gemm_final<<<dim3(O_P / 128, M_ROWS / 64), dim3(256), 0, stream>>>(hall, wfb, out, pS2, pQ2);
  // 6) fold final BN
  k_stats_final<<<dim3((O_O + 255) / 256), dim3(256), 0, stream>>>(pS2, pQ2, gf, bff, a2, c2);
  // 7) fused BN + log_softmax over each of the 64000 rows
  k_bn_logsoftmax<<<dim3(M_ROWS), dim3(256), 0, stream>>>(out, a2, c2);
}
